// VectorQuantizer_49684181680900
// MI455X (gfx1250) — matrix operand order repaired and checked by reading, not compiled, not run on hardware
//
#include <hip/hip_runtime.h>

typedef __attribute__((ext_vector_type(16))) _Float16 v16h;
typedef __attribute__((ext_vector_type(8)))  _Float16 v8h;
typedef __attribute__((ext_vector_type(8)))  float    v8f;

#define KCODES 1024
#define DDIM   64
#define NROWS  (32*64*64)        // 131072 rows
#define OUTEL  8388608           // 32*64*64*64

union AFrag { v16h v; v8h h[2]; };

__device__ __forceinline__ void argmin_take(float& bv, int& bi, float ov, int oi) {
    // branchless lexicographic (value, index) min
    bool take = (ov < bv) | ((ov == bv) & (oi < bi));
    bv = take ? ov : bv;
    bi = take ? oi : bi;
}

// ---------------- prep: f16 codebook (folded -2), ||e||^2, zero accumulators ----------------
__global__ __launch_bounds__(256) void vq_prep(const float* __restrict__ emb,
                                               _Float16* __restrict__ embh,
                                               float* __restrict__ e2,
                                               unsigned* __restrict__ hist,
                                               float* __restrict__ lossacc)
{
    int k = blockIdx.x * blockDim.x + threadIdx.x;
    if (k < KCODES) {
        float s = 0.f;
        #pragma unroll
        for (int c = 0; c < DDIM; ++c) {
            float v = emb[k * DDIM + c];
            s += v * v;
            embh[k * DDIM + c] = (_Float16)(-2.0f * v);
        }
        e2[k] = s;
        hist[k] = 0u;
        if (k == 0) *lossacc = 0.f;
    }
}

// ---------------- main: WMMA distances + argmin + gather + loss + histogram ----------------
__global__ __launch_bounds__(256) void vq_main(const float* __restrict__ x,
                                               const float* __restrict__ emb,
                                               const _Float16* __restrict__ embh,
                                               const float* __restrict__ e2,
                                               unsigned* __restrict__ hist,
                                               float* __restrict__ lossacc,
                                               float* __restrict__ zq_out)
{
    __shared__ _Float16 Ah[16 * 64];   // f16 A tile (rows x c), c contiguous
    __shared__ float    Af[16 * 64];   // f32 copy for loss
    __shared__ float    wval[8 * 16];  // per-wave row minima
    __shared__ int      widx[8 * 16];
    __shared__ int      fidx[16];
    __shared__ float    blsum;

    const int t   = threadIdx.x;
    const int m0  = blockIdx.x * 16;           // first row of this tile
    const int b   = m0 >> 12;                  // / (64*64)
    const int rem = m0 & 4095;
    const int h   = rem >> 6;
    const int w0  = rem & 63;                  // multiple of 16

    // ---- stage A tile: x[b, c, h, w0+r] ; 16 contiguous floats per (c) strip ----
    {
        int r  = t & 15;
        int c0 = t >> 4;                       // 0..15
        #pragma unroll
        for (int i = 0; i < 4; ++i) {
            int c = c0 + 16 * i;
            float v = x[b * 262144 + c * 4096 + h * 64 + w0 + r];
            Af[r * 64 + c] = v;
            Ah[r * 64 + c] = (_Float16)v;
        }
    }
    if (t == 0) blsum = 0.f;
    __syncthreads();

    const int wave = t >> 5;
    const int lane = t & 31;
    const int m    = lane & 15;
    const int hlf  = lane >> 4;                // 0/1 lane half
    const int kb8  = hlf * 8;                  // unused (A frags use kbB)
    const int kbB  = hlf * 16;                 // B-frag K sub-base

    // A fragments for K=0..31 and K=32..63 (contiguous K: k = 16*hlf + s, matching B)
    AFrag a0, a1;
    a0.h[0] = *(const v8h*)&Ah[m * 64 +  0 + kbB];
    a0.h[1] = *(const v8h*)&Ah[m * 64 +  8 + kbB];
    a1.h[0] = *(const v8h*)&Ah[m * 64 + 32 + kbB];
    a1.h[1] = *(const v8h*)&Ah[m * 64 + 40 + kbB];

    float minv[8];
    int   mini[8];
    #pragma unroll
    for (int v = 0; v < 8; ++v) { minv[v] = 3.4e38f; mini[v] = 0; }

    // each wave: 8 n-tiles of 16 codes; score = -2 z.e (WMMA, C seeded 0) + ||e||^2 (added late)
    #pragma unroll
    for (int j = 0; j < 8; ++j) {
        int n = wave * 128 + j * 16 + m;
        v16h b0 = *(const v16h*)&embh[n * 64 +  0 + kbB];
        v16h b1 = *(const v16h*)&embh[n * 64 + 32 + kbB];
        float e2n = e2[n];
        v8f acc = {};   // SRC2 = inline 0, no broadcast movs on the WMMA critical path
        acc = __builtin_amdgcn_wmma_f32_16x16x32_f16(false, a0.v, false, b0,
                                                     (short)0, acc, false, false);
        acc = __builtin_amdgcn_wmma_f32_16x16x32_f16(false, a1.v, false, b1,
                                                     (short)0, acc, false, false);
        #pragma unroll
        for (int v = 0; v < 8; ++v) {
            float d = acc[v] + e2n;
            bool lt = d < minv[v];             // ascending n: strict < keeps first min
            minv[v] = lt ? d : minv[v];
            mini[v] = lt ? n : mini[v];
        }
    }

    // ---- cross-lane argmin within each 16-lane half (ds_swizzle XOR butterfly) ----
    #pragma unroll
    for (int v = 0; v < 8; ++v) {
        float bv = minv[v]; int bi = mini[v];
        { float ov = __int_as_float(__builtin_amdgcn_ds_swizzle(__float_as_int(bv), 0x041F));
          int   oi = __builtin_amdgcn_ds_swizzle(bi, 0x041F);
          argmin_take(bv, bi, ov, oi); }
        { float ov = __int_as_float(__builtin_amdgcn_ds_swizzle(__float_as_int(bv), 0x081F));
          int   oi = __builtin_amdgcn_ds_swizzle(bi, 0x081F);
          argmin_take(bv, bi, ov, oi); }
        { float ov = __int_as_float(__builtin_amdgcn_ds_swizzle(__float_as_int(bv), 0x101F));
          int   oi = __builtin_amdgcn_ds_swizzle(bi, 0x101F);
          argmin_take(bv, bi, ov, oi); }
        { float ov = __int_as_float(__builtin_amdgcn_ds_swizzle(__float_as_int(bv), 0x201F));
          int   oi = __builtin_amdgcn_ds_swizzle(bi, 0x201F);
          argmin_take(bv, bi, ov, oi); }
        if (m == 0) {
            int row = v + 8 * hlf;             // C/D layout: m = v + 8*(lane/16)
            wval[wave * 16 + row] = bv;
            widx[wave * 16 + row] = bi;
        }
    }
    __syncthreads();

    // ---- cross-wave reduce (waves cover ascending n ranges) + histogram ----
    if (t < 16) {
        float bv = wval[t]; int bi = widx[t];
        #pragma unroll
        for (int w = 1; w < 8; ++w) {
            argmin_take(bv, bi, wval[w * 16 + t], widx[w * 16 + t]);
        }
        fidx[t] = bi;
        atomicAdd(&hist[bi], 1u);
    }
    __syncthreads();

    // ---- gather z_q, write [B,D,H,W] (coalesced over w), accumulate loss ----
    float lsum = 0.f;
    {
        int r  = t & 15;
        int d0 = t >> 4;
        int idx = fidx[r];
        const int obase = b * 262144 + h * 64 + w0 + r;
        #pragma unroll
        for (int i = 0; i < 4; ++i) {
            int d = d0 + 16 * i;
            float q = emb[idx * 64 + d];
            zq_out[obase + d * 4096] = q;
            float diff = q - Af[r * 64 + d];
            lsum += diff * diff;
        }
    }
    atomicAdd(&blsum, lsum);
    __syncthreads();
    if (t == 0) atomicAdd(lossacc, blsum);
}

// ---------------- finalize: loss scale + perplexity ----------------
__global__ __launch_bounds__(256) void vq_final(const unsigned* __restrict__ hist,
                                                const float* __restrict__ lossacc,
                                                float* __restrict__ out)
{
    __shared__ float red[256];
    int t = threadIdx.x;
    float s = 0.f;
    for (int k = t; k < KCODES; k += 256) {
        float p = (float)hist[k] * (1.0f / (float)NROWS);
        s += p * logf(p + 1e-10f);
    }
    red[t] = s;
    __syncthreads();
    for (int off = 128; off > 0; off >>= 1) {
        if (t < off) red[t] += red[t + off];
        __syncthreads();
    }
    if (t == 0) {
        out[0]         = 1.25f * (*lossacc) * (1.0f / (float)OUTEL);  // (1+beta)*MSE
        out[OUTEL + 1] = expf(-red[0]);
    }
}

extern "C" void kernel_launch(void* const* d_in, const int* in_sizes, int n_in,
                              void* d_out, int out_size, void* d_ws, size_t ws_size,
                              hipStream_t stream)
{
    const float* x   = (const float*)d_in[0];   // [32,64,64,64] f32
    const float* emb = (const float*)d_in[1];   // [1024,64] f32

    char* ws = (char*)d_ws;
    _Float16* embh   = (_Float16*)ws;                      // 131072 B (64B-aligned base)
    float*    e2     = (float*)(ws + 131072);              // 4096 B
    unsigned* hist   = (unsigned*)(ws + 131072 + 4096);    // 4096 B
    float*    lossacc= (float*)(ws + 131072 + 8192);       // 4 B

    float* out = (float*)d_out;   // [loss | z_q(8388608) | perplexity]

    vq_prep<<<4, 256, 0, stream>>>(emb, embh, e2, hist, lossacc);
    vq_main<<<NROWS / 16, 256, 0, stream>>>(x, emb, embh, e2, hist, lossacc, out + 1);
    vq_final<<<1, 256, 0, stream>>>(hist, lossacc, out);
}
